// BondAwareGNNWithCoordUpdate_50491635531848
// MI455X (gfx1250) — compile-verified, matrix operand order checked by reading, not run on hardware
//
#include <hip/hip_runtime.h>

typedef __attribute__((ext_vector_type(16))) _Float16 v16h;
typedef __attribute__((ext_vector_type(8)))  _Float16 v8h;
typedef __attribute__((ext_vector_type(8)))  float    v8f;

__device__ __forceinline__ float silu_f(float x) {
  return x * (1.0f / (1.0f + __expf(-x)));
}

__device__ __forceinline__ v8f wmma_f16(v16h a, v16h b, v8f c) {
  // D = A(16x32,f16) x B(32x16,f16) + C(16x16,f32)
  return __builtin_amdgcn_wmma_f32_16x16x32_f16(false, a, false, b, (short)0, c,
                                                false, false);
}

// A-fragment loader from LDS (row-major [rows][stride_h] halfs).
// lane L: row = L%16 ; 16 halfs at K = k0 + e + (e>=8?8:0) + (L>=16?8:0)
// => two contiguous 8-half (16B) runs: [k0+off, +8) and [k0+off+16, +8)
__device__ __forceinline__ v16h load_a_frag(const _Float16* A, int stride_h,
                                            int lane, int k0) {
  const int r = lane & 15;
  const int off = (lane >> 4) << 3;
  const _Float16* p = A + r * stride_h + k0 + off;
  union { v16h v; v8h h[2]; } u;
  u.h[0] = *(const v8h*)(p);
  u.h[1] = *(const v8h*)(p + 16);
  return u.v;
}

// CDNA5 async global->LDS copy (16B per lane), tracked on ASYNCcnt.
__device__ __forceinline__ void async_load_b128(unsigned lds_off, const void* gaddr) {
  asm volatile("global_load_async_to_lds_b128 %0, %1, off"
               :: "v"(lds_off), "v"(gaddr) : "memory");
}
__device__ __forceinline__ void wait_asynccnt0() {
  asm volatile("s_wait_asynccnt 0x0" ::: "memory");
}
__device__ __forceinline__ unsigned lds_offset_of(const void* p) {
  return (unsigned)(unsigned long long)p;  // LDS generic addr: low 32 bits = LDS offset
}

// -------------------- weight pre-swizzle (B-fragment layout) --------------------
// packed[((kt*ntiles + nt)*32 + lane)*16 + e] = W[K][nt*16 + lane%16] as f16,
// K = kt*32 + e + (e>=8?8:0) + (lane>=16?8:0)
__global__ void pack_frag_kernel(const float* __restrict__ W, _Float16* __restrict__ out,
                                 int K, int Ncols) {
  const int ntiles = Ncols / 16;
  const int total = K * Ncols;
  int idx = blockIdx.x * blockDim.x + threadIdx.x;
  if (idx >= total) return;
  int kt  = idx / (ntiles * 512);
  int rem = idx % (ntiles * 512);
  int nt  = rem / 512;
  int r2  = rem % 512;
  int lane = r2 / 16;
  int e    = r2 % 16;
  int Kidx = kt * 32 + e + ((e >= 8) ? 8 : 0) + ((lane >= 16) ? 8 : 0);
  int Nidx = nt * 16 + (lane & 15);
  out[idx] = (_Float16)W[(size_t)Kidx * Ncols + Nidx];
}

__global__ void wm1_last_kernel(const float* __restrict__ Wm1, float* __restrict__ last,
                                int L, int H, int K1) {
  int t = blockIdx.x * blockDim.x + threadIdx.x;
  if (t >= L * H) return;
  int l = t / H, n = t % H;
  last[t] = Wm1[((size_t)l * K1 + (K1 - 1)) * H + n];
}

// -------------------- tiny setup kernels --------------------
__global__ void time_emb_kernel(const float* __restrict__ t,
                                const float* __restrict__ W_t1, const float* __restrict__ b_t1,
                                const float* __restrict__ W_t2, const float* __restrict__ b_t2,
                                float* __restrict__ t_emb, int B, int T, int H) {
  __shared__ float s_te[256];
  __shared__ float s_hid[256];
  const int tid = threadIdx.x;
  const int half = T / 2;
  for (int b = 0; b < B; ++b) {
    if (tid < T) {
      int j = tid % half;
      float fr = __expf((float)j * (-logf(10000.0f) / (float)(half - 1)));
      float v = t[b] * fr;
      s_te[tid] = (tid < half) ? sinf(v) : cosf(v);
    }
    __syncthreads();
    float a = b_t1[tid];
    for (int j = 0; j < T; ++j) a += s_te[j] * W_t1[j * H + tid];
    s_hid[tid] = silu_f(a);
    __syncthreads();
    float o = b_t2[tid];
    for (int k = 0; k < H; ++k) o += s_hid[k] * W_t2[k * H + tid];
    t_emb[b * H + tid] = o;
    __syncthreads();
  }
}

__global__ void htype_kernel(const float* __restrict__ emb, const float* __restrict__ Wn,
                             const float* __restrict__ bn, float* __restrict__ htype,
                             int A, int H) {
  int tt = blockIdx.x * blockDim.x + threadIdx.x;
  if (tt >= A * H) return;
  int a = tt / H, c = tt % H;
  float s = bn[c];
  for (int k = 0; k < H; ++k) s += emb[a * H + k] * Wn[k * H + c];
  htype[tt] = s;
}

__global__ void h_init_kernel(const int* __restrict__ atype, const float* __restrict__ htype,
                              const float* __restrict__ t_emb, float* __restrict__ h,
                              _Float16* __restrict__ h_half, int N, int H, long total) {
  long i = (long)blockIdx.x * blockDim.x + threadIdx.x;
  if (i >= total) return;
  int c = (int)(i % H);
  long rem = i / H;
  int nn = (int)(rem % N);
  int b = (int)(rem / N);
  float v = htype[atype[nn] * H + c] + t_emb[b * H + c];
  h[i] = v;
  h_half[i] = (_Float16)v;
}

__global__ void zero_f32_kernel(float* __restrict__ pz, long n) {
  long i = (long)blockIdx.x * blockDim.x + threadIdx.x;
  if (i < n) pz[i] = 0.0f;
}

__global__ void copy_f32_kernel(float* __restrict__ dst, const float* __restrict__ src, long n) {
  long i = (long)blockIdx.x * blockDim.x + threadIdx.x;
  if (i < n) dst[i] = src[i];
}

__global__ void count_kernel(const int* __restrict__ bond, float* __restrict__ counts, int E) {
  int e = blockIdx.x * blockDim.x + threadIdx.x;
  if (e >= E) return;
  atomicAdd(&counts[bond[2 * e + 0]], 1.0f);
  atomicAdd(&counts[bond[2 * e + 1]], 1.0f);
}

__global__ void cinv_kernel(float* __restrict__ counts, int N) {
  int i = blockIdx.x * blockDim.x + threadIdx.x;
  if (i < N) counts[i] = 1.0f / fmaxf(counts[i], 1.0f);
}

// -------------------- fused edge kernel (per layer) --------------------
// tile of 32 edges x batch b; 8 waves; wave = N-tile, 2 M-tiles per wave.
__global__ __launch_bounds__(256) void edge_kernel(
    const float* __restrict__ x_cur, float* __restrict__ x_next,
    const _Float16* __restrict__ h_half, float* __restrict__ agg,
    const int* __restrict__ bond, const float* __restrict__ cinv,
    const v16h* __restrict__ pkWm1, const float* __restrict__ Wm1_last,
    const float* __restrict__ bm1,
    const v16h* __restrict__ pkWm2, const float* __restrict__ bm2,
    const v16h* __restrict__ pkWc1, const float* __restrict__ bc1,
    const float* __restrict__ Wc2,
    int N, int H)
{
  __shared__ int s_src[32], s_dst[32];
  __shared__ float s_dist[32], s_cw[32];
  __shared__ float s_cd[32][3];
  __shared__ __align__(16) _Float16 sA[32][256];   // [h_src | h_dst] f16
  __shared__ __align__(16) _Float16 sM1[32][128];  // silu(msg_in@Wm1)
  __shared__ __align__(16) float    sMf[32][128];  // m (f32, for atomics)
  __shared__ __align__(16) _Float16 sMh[32][128];  // m (f16, GEMM3 operand)

  const int tid  = threadIdx.x;
  const int lane = tid & 31;
  const int wave = tid >> 5;
  const int b    = blockIdx.y;
  const long bn = (long)b * N;

  if (tid < 32) {
    int e = blockIdx.x * 32 + tid;
    int s = bond[2 * e + 0];
    int d = bond[2 * e + 1];
    s_src[tid] = s; s_dst[tid] = d;
    float dx = x_cur[(bn + d) * 3 + 0] - x_cur[(bn + s) * 3 + 0];
    float dy = x_cur[(bn + d) * 3 + 1] - x_cur[(bn + s) * 3 + 1];
    float dz = x_cur[(bn + d) * 3 + 2] - x_cur[(bn + s) * 3 + 2];
    s_cd[tid][0] = dx; s_cd[tid][1] = dy; s_cd[tid][2] = dz;
    s_dist[tid] = sqrtf(dx * dx + dy * dy + dz * dz);
    s_cw[tid] = 0.0f;
  }
  __syncthreads();

  // stage A = [h_src | h_dst] (32 rows x 256 halfs) via async global->LDS
  {
    int r = tid >> 3;
    int c0 = (tid & 7) * 32;  // 32 halfs = 64 bytes = 4 x b128
    const _Float16* srcp = (c0 < 128)
        ? (h_half + (bn + s_src[r]) * H + c0)
        : (h_half + (bn + s_dst[r]) * H + (c0 - 128));
    unsigned lo = lds_offset_of(&sA[r][c0]);
    #pragma unroll
    for (int i = 0; i < 4; ++i)
      async_load_b128(lo + 16u * i, srcp + 8 * i);
  }
  wait_asynccnt0();
  __syncthreads();

  const int n = wave * 16 + (lane & 15);
  const int roff = (lane >> 4) * 8;

  // GEMM1: msg_in(32x257) @ Wm1 -> silu -> sM1
  {
    v8f acc0, acc1;
    float bv = bm1[n];
    #pragma unroll
    for (int i = 0; i < 8; ++i) { acc0[i] = bv; acc1[i] = bv; }
    #pragma unroll
    for (int kt = 0; kt < 8; ++kt) {
      v16h bf = pkWm1[(kt * 8 + wave) * 32 + lane];
      v16h a0 = load_a_frag(&sA[0][0],  256, lane, kt * 32);
      v16h a1 = load_a_frag(&sA[16][0], 256, lane, kt * 32);
      acc0 = wmma_f16(a0, bf, acc0);
      acc1 = wmma_f16(a1, bf, acc1);
    }
    float wl = Wm1_last[n];
    #pragma unroll
    for (int r = 0; r < 8; ++r) {
      int r0 = r + roff, r1 = r0 + 16;
      sM1[r0][n] = (_Float16)silu_f(acc0[r] + s_dist[r0] * wl);  // 'dist' column
      sM1[r1][n] = (_Float16)silu_f(acc1[r] + s_dist[r1] * wl);
    }
  }
  __syncthreads();

  // GEMM2: sM1 @ Wm2 -> silu -> m
  {
    v8f acc0, acc1;
    float bv = bm2[n];
    #pragma unroll
    for (int i = 0; i < 8; ++i) { acc0[i] = bv; acc1[i] = bv; }
    #pragma unroll
    for (int kt = 0; kt < 4; ++kt) {
      v16h bf = pkWm2[(kt * 8 + wave) * 32 + lane];
      v16h a0 = load_a_frag(&sM1[0][0],  128, lane, kt * 32);
      v16h a1 = load_a_frag(&sM1[16][0], 128, lane, kt * 32);
      acc0 = wmma_f16(a0, bf, acc0);
      acc1 = wmma_f16(a1, bf, acc1);
    }
    #pragma unroll
    for (int r = 0; r < 8; ++r) {
      int r0 = r + roff, r1 = r0 + 16;
      float v0 = silu_f(acc0[r]);
      float v1 = silu_f(acc1[r]);
      sMf[r0][n] = v0; sMh[r0][n] = (_Float16)v0;
      sMf[r1][n] = v1; sMh[r1][n] = (_Float16)v1;
    }
  }
  __syncthreads();

  // GEMM3: m @ Wc1 -> silu -> dot Wc2 -> cw per edge (ds_add reduction)
  {
    v8f acc0, acc1;
    float bv = bc1[n];
    #pragma unroll
    for (int i = 0; i < 8; ++i) { acc0[i] = bv; acc1[i] = bv; }
    #pragma unroll
    for (int kt = 0; kt < 4; ++kt) {
      v16h bf = pkWc1[(kt * 8 + wave) * 32 + lane];
      v16h a0 = load_a_frag(&sMh[0][0],  128, lane, kt * 32);
      v16h a1 = load_a_frag(&sMh[16][0], 128, lane, kt * 32);
      acc0 = wmma_f16(a0, bf, acc0);
      acc1 = wmma_f16(a1, bf, acc1);
    }
    float w2 = Wc2[n];
    #pragma unroll
    for (int r = 0; r < 8; ++r) {
      atomicAdd(&s_cw[r + roff],      silu_f(acc0[r]) * w2);
      atomicAdd(&s_cw[r + roff + 16], silu_f(acc1[r]) * w2);
    }
  }
  __syncthreads();

  // coordinate scatter: x_next += (+-)cdn*cw / counts
  if (tid < 96) {
    int r = tid / 3, c = tid % 3;
    float cw = s_cw[r];
    float dist = s_dist[r];
    float upd = (s_cd[r][c] / (dist + 1e-8f)) * cw;
    int d = s_dst[r], s = s_src[r];
    atomicAdd(&x_next[(bn + d) * 3 + c],  upd * cinv[d]);
    atomicAdd(&x_next[(bn + s) * 3 + c], -upd * cinv[s]);
  }

  // agg scatter: agg[dst] += m ; agg[src] += m
  for (int idx = tid; idx < 32 * 128; idx += 256) {
    int r = idx >> 7, c = idx & 127;
    float v = sMf[r][c];
    atomicAdd(&agg[(bn + s_dst[r]) * H + c], v);
    atomicAdd(&agg[(bn + s_src[r]) * H + c], v);
  }
}

// -------------------- node update kernel (per layer) --------------------
__global__ __launch_bounds__(256) void node_kernel(
    float* __restrict__ h, _Float16* __restrict__ h_half,
    const float* __restrict__ agg,
    const v16h* __restrict__ pkWn1, const float* __restrict__ bn1,
    const v16h* __restrict__ pkWn2, const float* __restrict__ bn2,
    int N, int H)
{
  __shared__ __align__(16) _Float16 sA[32][256];  // [h | agg]
  __shared__ __align__(16) _Float16 sT[32][128];

  const int tid = threadIdx.x;
  const int lane = tid & 31;
  const int wave = tid >> 5;
  const long base = (long)blockIdx.y * N + (long)blockIdx.x * 32;

  {
    int r = tid >> 3;
    int c0 = (tid & 7) * 32;
    if (c0 < 128) {
      const _Float16* srcp = h_half + (base + r) * H + c0;
      unsigned lo = lds_offset_of(&sA[r][c0]);
      #pragma unroll
      for (int i = 0; i < 4; ++i)
        async_load_b128(lo + 16u * i, srcp + 8 * i);
    } else {
      const float* ap = agg + (base + r) * H + (c0 - 128);
      #pragma unroll
      for (int i = 0; i < 32; ++i) sA[r][c0 + i] = (_Float16)ap[i];
    }
  }
  wait_asynccnt0();
  __syncthreads();

  const int n = wave * 16 + (lane & 15);
  const int roff = (lane >> 4) * 8;

  {
    v8f acc0, acc1;
    float bv = bn1[n];
    #pragma unroll
    for (int i = 0; i < 8; ++i) { acc0[i] = bv; acc1[i] = bv; }
    #pragma unroll
    for (int kt = 0; kt < 8; ++kt) {
      v16h bf = pkWn1[(kt * 8 + wave) * 32 + lane];
      v16h a0 = load_a_frag(&sA[0][0],  256, lane, kt * 32);
      v16h a1 = load_a_frag(&sA[16][0], 256, lane, kt * 32);
      acc0 = wmma_f16(a0, bf, acc0);
      acc1 = wmma_f16(a1, bf, acc1);
    }
    #pragma unroll
    for (int r = 0; r < 8; ++r) {
      sT[r + roff][n]      = (_Float16)silu_f(acc0[r]);
      sT[r + roff + 16][n] = (_Float16)silu_f(acc1[r]);
    }
  }
  __syncthreads();

  {
    v8f acc0, acc1;
    float bv = bn2[n];
    #pragma unroll
    for (int i = 0; i < 8; ++i) { acc0[i] = bv; acc1[i] = bv; }
    #pragma unroll
    for (int kt = 0; kt < 4; ++kt) {
      v16h bf = pkWn2[(kt * 8 + wave) * 32 + lane];
      v16h a0 = load_a_frag(&sT[0][0],  128, lane, kt * 32);
      v16h a1 = load_a_frag(&sT[16][0], 128, lane, kt * 32);
      acc0 = wmma_f16(a0, bf, acc0);
      acc1 = wmma_f16(a1, bf, acc1);
    }
    #pragma unroll
    for (int r = 0; r < 8; ++r) {
      long i0 = (base + r + roff) * H + n;
      long i1 = (base + r + roff + 16) * H + n;
      float v0 = h[i0] + acc0[r];
      float v1 = h[i1] + acc1[r];
      h[i0] = v0; h_half[i0] = (_Float16)v0;
      h[i1] = v1; h_half[i1] = (_Float16)v1;
    }
  }
}

// -------------------- output kernel --------------------
__global__ __launch_bounds__(256) void output_kernel(
    const _Float16* __restrict__ h_half,
    const float* __restrict__ x_fin, const float* __restrict__ x_orig,
    const v16h* __restrict__ pkWo1, const float* __restrict__ b_o1,
    const float* __restrict__ W_o2, const float* __restrict__ b_o2,
    float* __restrict__ out, int N, int H)
{
  __shared__ __align__(16) _Float16 sA[32][128];
  __shared__ float sS[32][128];

  const int tid = threadIdx.x;
  const int lane = tid & 31;
  const int wave = tid >> 5;
  const long base = (long)blockIdx.y * N + (long)blockIdx.x * 32;

  {
    int r = tid >> 3;
    int c0 = (tid & 7) * 16;  // 16 halfs = 32B = 2 x b128
    const _Float16* srcp = h_half + (base + r) * H + c0;
    unsigned lo = lds_offset_of(&sA[r][c0]);
    async_load_b128(lo,       srcp);
    async_load_b128(lo + 16u, srcp + 8);
  }
  wait_asynccnt0();
  __syncthreads();

  const int n = wave * 16 + (lane & 15);
  const int roff = (lane >> 4) * 8;

  v8f acc0, acc1;
  float bv = b_o1[n];
  #pragma unroll
  for (int i = 0; i < 8; ++i) { acc0[i] = bv; acc1[i] = bv; }
  #pragma unroll
  for (int kt = 0; kt < 4; ++kt) {
    v16h bf = pkWo1[(kt * 8 + wave) * 32 + lane];
    v16h a0 = load_a_frag(&sA[0][0],  128, lane, kt * 32);
    v16h a1 = load_a_frag(&sA[16][0], 128, lane, kt * 32);
    acc0 = wmma_f16(a0, bf, acc0);
    acc1 = wmma_f16(a1, bf, acc1);
  }
  #pragma unroll
  for (int r = 0; r < 8; ++r) {
    sS[r + roff][n]      = silu_f(acc0[r]);
    sS[r + roff + 16][n] = silu_f(acc1[r]);
  }
  __syncthreads();

  if (tid < 96) {
    int r = tid / 3, c = tid % 3;
    float s = b_o2[c];
    for (int k = 0; k < 128; ++k) s += sS[r][k] * W_o2[k * 3 + c];
    long idx = (base + r) * 3 + c;
    out[idx] = x_fin[idx] - x_orig[idx] + s;
  }
}

// -------------------- host orchestration --------------------
extern "C" void kernel_launch(void* const* d_in, const int* in_sizes, int n_in,
                              void* d_out, int out_size, void* d_ws, size_t ws_size,
                              hipStream_t stream) {
  (void)n_in; (void)out_size; (void)ws_size;
  const float* x      = (const float*)d_in[0];
  const float* t      = (const float*)d_in[1];
  const int*   atype  = (const int*)d_in[2];
  const int*   bond   = (const int*)d_in[3];
  const float* emb    = (const float*)d_in[4];
  const float* W_t1   = (const float*)d_in[5];
  const float* b_t1   = (const float*)d_in[6];
  const float* W_t2   = (const float*)d_in[7];
  const float* b_t2   = (const float*)d_in[8];
  const float* W_node = (const float*)d_in[9];
  const float* b_node = (const float*)d_in[10];
  const float* Wm1    = (const float*)d_in[11];
  const float* bm1    = (const float*)d_in[12];
  const float* Wm2    = (const float*)d_in[13];
  const float* bm2    = (const float*)d_in[14];
  const float* Wn1    = (const float*)d_in[15];
  const float* bn1    = (const float*)d_in[16];
  const float* Wn2    = (const float*)d_in[17];
  const float* bn2    = (const float*)d_in[18];
  const float* Wc1    = (const float*)d_in[19];
  const float* bc1    = (const float*)d_in[20];
  const float* Wc2    = (const float*)d_in[21];
  const float* W_o1   = (const float*)d_in[22];
  const float* b_o1   = (const float*)d_in[23];
  const float* W_o2   = (const float*)d_in[24];
  const float* b_o2   = (const float*)d_in[25];
  float* out = (float*)d_out;

  const int B = in_sizes[1];
  const int N = in_sizes[2];
  const int E = in_sizes[3] / 2;
  const int H = in_sizes[6];           // 128
  const int T = in_sizes[5] / H;       // 64
  const int A = in_sizes[4] / H;       // 4
  const int L = in_sizes[12] / H;      // 4
  const int K1 = 2 * H + 1;            // 257
  const long BNH = (long)B * N * H;
  const long BN3 = (long)B * N * 3;

  char* p = (char*)d_ws;
  auto walloc = [&](size_t bytes) -> char* {
    char* r = p; p += (bytes + 255) & ~(size_t)255; return r;
  };
  float*    t_emb   = (float*)walloc((size_t)B * H * 4);
  float*    htypep  = (float*)walloc((size_t)A * H * 4);
  float*    cinv    = (float*)walloc((size_t)N * 4);
  float*    x_cur   = (float*)walloc((size_t)BN3 * 4);
  float*    x_next  = (float*)walloc((size_t)BN3 * 4);
  float*    h       = (float*)walloc((size_t)BNH * 4);
  _Float16* h_half  = (_Float16*)walloc((size_t)BNH * 2);
  float*    agg     = (float*)walloc((size_t)BNH * 4);
  _Float16* pk_m1   = (_Float16*)walloc((size_t)L * 2 * H * H * 2);
  float*    m1_last = (float*)walloc((size_t)L * H * 4);
  _Float16* pk_m2   = (_Float16*)walloc((size_t)L * H * H * 2);
  _Float16* pk_n1   = (_Float16*)walloc((size_t)L * 2 * H * H * 2);
  _Float16* pk_n2   = (_Float16*)walloc((size_t)L * H * H * 2);
  _Float16* pk_c1   = (_Float16*)walloc((size_t)L * H * H * 2);
  _Float16* pk_o1   = (_Float16*)walloc((size_t)H * H * 2);

  const dim3 blk(256);

  // weight pre-swizzle
  for (int l = 0; l < L; ++l) {
    pack_frag_kernel<<<(2 * H * H + 255) / 256, blk, 0, stream>>>(
        Wm1 + (size_t)l * K1 * H, pk_m1 + (size_t)l * 2 * H * H, 2 * H, H);
    pack_frag_kernel<<<(H * H + 255) / 256, blk, 0, stream>>>(
        Wm2 + (size_t)l * H * H, pk_m2 + (size_t)l * H * H, H, H);
    pack_frag_kernel<<<(2 * H * H + 255) / 256, blk, 0, stream>>>(
        Wn1 + (size_t)l * 2 * H * H, pk_n1 + (size_t)l * 2 * H * H, 2 * H, H);
    pack_frag_kernel<<<(H * H + 255) / 256, blk, 0, stream>>>(
        Wn2 + (size_t)l * H * H, pk_n2 + (size_t)l * H * H, H, H);
    pack_frag_kernel<<<(H * H + 255) / 256, blk, 0, stream>>>(
        Wc1 + (size_t)l * H * H, pk_c1 + (size_t)l * H * H, H, H);
  }
  pack_frag_kernel<<<(H * H + 255) / 256, blk, 0, stream>>>(W_o1, pk_o1, H, H);
  wm1_last_kernel<<<(L * H + 255) / 256, blk, 0, stream>>>(Wm1, m1_last, L, H, K1);

  // setup
  time_emb_kernel<<<1, H, 0, stream>>>(t, W_t1, b_t1, W_t2, b_t2, t_emb, B, T, H);
  htype_kernel<<<(A * H + 255) / 256, blk, 0, stream>>>(emb, W_node, b_node, htypep, A, H);
  zero_f32_kernel<<<(N + 255) / 256, blk, 0, stream>>>(cinv, N);
  count_kernel<<<(E + 255) / 256, blk, 0, stream>>>(bond, cinv, E);
  cinv_kernel<<<(N + 255) / 256, blk, 0, stream>>>(cinv, N);
  h_init_kernel<<<(int)((BNH + 255) / 256), blk, 0, stream>>>(atype, htypep, t_emb, h, h_half, N, H, BNH);
  copy_f32_kernel<<<(int)((BN3 + 255) / 256), blk, 0, stream>>>(x_cur, x, BN3);

  float* xc = x_cur;
  float* xn = x_next;
  for (int l = 0; l < L; ++l) {
    zero_f32_kernel<<<(int)((BNH + 255) / 256), blk, 0, stream>>>(agg, BNH);
    copy_f32_kernel<<<(int)((BN3 + 255) / 256), blk, 0, stream>>>(xn, xc, BN3);
    edge_kernel<<<dim3(E / 32, B), blk, 0, stream>>>(
        xc, xn, h_half, agg, bond, cinv,
        (const v16h*)(pk_m1 + (size_t)l * 2 * H * H), m1_last + (size_t)l * H,
        bm1 + (size_t)l * H,
        (const v16h*)(pk_m2 + (size_t)l * H * H), bm2 + (size_t)l * H,
        (const v16h*)(pk_c1 + (size_t)l * H * H), bc1 + (size_t)l * H,
        Wc2 + (size_t)l * H, N, H);
    node_kernel<<<dim3(N / 32, B), blk, 0, stream>>>(
        h, h_half, agg,
        (const v16h*)(pk_n1 + (size_t)l * 2 * H * H), bn1 + (size_t)l * H,
        (const v16h*)(pk_n2 + (size_t)l * H * H), bn2 + (size_t)l * H, N, H);
    float* tmp = xc; xc = xn; xn = tmp;
  }

  output_kernel<<<dim3(N / 32, B), blk, 0, stream>>>(
      h_half, xc, x, (const v16h*)pk_o1, b_o1, W_o2, b_o2, out, N, H);
}
